// StressNet_15642270892631
// MI455X (gfx1250) — matrix operand order repaired and checked by reading, not compiled, not run on hardware
//
#include <hip/hip_runtime.h>
#include <hip/hip_fp16.h>

// ---------------------------------------------------------------------------
// CDNA5 (gfx1250) implementation of the PointConv-style StressNet reference.
// wave32; WMMA 16x16x4 (f32) for density gram matrices, WMMA 16x16x32 (f16,
// f32-accum) for channel GEMMs / einsum with f16-prepacked operands, and
// async global->LDS staging (ASYNCcnt) for shared weight tiles.
// ---------------------------------------------------------------------------

typedef __attribute__((ext_vector_type(2)))  float    v2f;
typedef __attribute__((ext_vector_type(8)))  float    v8f;
typedef __attribute__((ext_vector_type(8)))  _Float16 v8h;
typedef __attribute__((ext_vector_type(16))) _Float16 v16h;
typedef int v4i __attribute__((vector_size(16)));   // b128 payload type

#if defined(__has_builtin)
#if __has_builtin(__builtin_amdgcn_global_load_async_to_lds_b128) && \
    __has_builtin(__builtin_amdgcn_s_wait_asynccnt)
#define HAVE_ASYNC 1
#endif
#endif
#ifndef HAVE_ASYNC
#define HAVE_ASYNC 0
#endif

static constexpr int BATCH = 8;
static constexpr int N1 = 4096, NP1 = 512, NS1 = 32, D0 = 5, KP1 = 32,  C1 = 64;
static constexpr int N2 = 512,  NP2 = 128, NS2 = 64, KP2 = 96, C2 = 128;
static constexpr int N3 = 128,  KP3 = 160, C3 = 256;   // ceil32(3+128)=160

// ---------------------------------------------------------------------------
// K0: pc (B,5,N) -> xyz (B,N,3), pts (B,N,5)
// ---------------------------------------------------------------------------
__global__ void k_transpose(const float* __restrict__ pc, float* __restrict__ xyz,
                            float* __restrict__ pts, int C, int N) {
  int i = blockIdx.x * blockDim.x + threadIdx.x;
  if (i >= BATCH * N) return;
  int b = i / N, n = i % N;
  for (int c = 0; c < C; ++c) {
    float v = pc[((size_t)b * C + c) * N + n];
    pts[((size_t)b * N + n) * C + c] = v;
    if (c < 3) xyz[((size_t)b * N + n) * 3 + c] = v;
  }
}

// ---------------------------------------------------------------------------
// K1: inverse Gaussian-KDE density via V_WMMA_F32_16X16X4_F32 (K pad 3->4).
// ---------------------------------------------------------------------------
__global__ void k_density(const float* __restrict__ xyz, float* __restrict__ invden,
                          int N, float bw) {
  const int b = blockIdx.y;
  const int row0 = blockIdx.x << 4;
  const int t = threadIdx.x;               // 256 threads = 8 waves
  const int wave = t >> 5, lane = t & 31;
  __shared__ float s_rowsq[16];
  __shared__ float s_sum[16];
  if (t < 16) {
    const float* p = xyz + ((size_t)b * N + row0 + t) * 3;
    s_rowsq[t] = p[0]*p[0] + p[1]*p[1] + p[2]*p[2];
    s_sum[t] = 0.f;
  }
  __syncthreads();
  v2f a;
  {
    int m = row0 + (lane & 15);
    const float* p = xyz + ((size_t)b * N + m) * 3;
    if (lane < 16) { a[0] = p[0]; a[1] = p[1]; }
    else           { a[0] = p[2]; a[1] = 0.f; }
  }
  const float inv2bw2 = 1.f / (2.f * bw * bw);
  float accum[8];
#pragma unroll
  for (int g = 0; g < 8; ++g) accum[g] = 0.f;
  const int tiles = N >> 4;
  for (int ct = wave; ct < tiles; ct += 8) {
    int n = (ct << 4) + (lane & 15);
    const float* q = xyz + ((size_t)b * N + n) * 3;
    v2f bf;
    if (lane < 16) { bf[0] = q[0]; bf[1] = q[2]; }
    else           { bf[0] = q[1]; bf[1] = 0.f; }
    float csq = q[0]*q[0] + q[1]*q[1] + q[2]*q[2];
    v8f c = {};
    c = __builtin_amdgcn_wmma_f32_16x16x4_f32(false, a, false, bf, (short)0, c,
                                              false, false);
#pragma unroll
    for (int g = 0; g < 8; ++g) {
      int r = g + ((lane < 16) ? 0 : 8);
      float d2 = s_rowsq[r] + csq - 2.f * c[g];
      accum[g] += __expf(-d2 * inv2bw2);
    }
  }
#pragma unroll
  for (int g = 0; g < 8; ++g) {
    int r = g + ((lane < 16) ? 0 : 8);
    atomicAdd(&s_sum[r], accum[g]);
  }
  __syncthreads();
  if (t < 16) {
    float s = s_sum[t] / (2.5f * bw);
    invden[(size_t)b * N + row0 + t] = (float)N / fmaxf(s, 1e-20f);
  }
}

// ---------------------------------------------------------------------------
// K2: farthest-point sampling (serial scan, register-resident points).
// ---------------------------------------------------------------------------
template<int N, int NP>
__global__ void k_fps(const float* __restrict__ xyz, int* __restrict__ out) {
  const int b = blockIdx.x, t = threadIdx.x;
  constexpr int T = 256, P = N / T;
  __shared__ float sv[T]; __shared__ int si[T];
  __shared__ int s_far;
  float px[P], py[P], pz[P], d[P];
#pragma unroll
  for (int i = 0; i < P; ++i) {
    int p = i * T + t;
    const float* q = xyz + ((size_t)b * N + p) * 3;
    px[i] = q[0]; py[i] = q[1]; pz[i] = q[2];
    d[i] = 1e10f;
  }
  if (t == 0) s_far = 0;
  __syncthreads();
  for (int it = 0; it < NP; ++it) {
    int f = s_far;
    if (t == 0) out[b * NP + it] = f;
    const float* q = xyz + ((size_t)b * N + f) * 3;
    float cx = q[0], cy = q[1], cz = q[2];
    float bm = -1.f; int bi = 0;
#pragma unroll
    for (int i = 0; i < P; ++i) {
      float dx = px[i] - cx, dy = py[i] - cy, dz = pz[i] - cz;
      float dd = dx*dx + dy*dy + dz*dz;
      d[i] = fminf(d[i], dd);
      if (d[i] > bm) { bm = d[i]; bi = i * T + t; }
    }
    sv[t] = bm; si[t] = bi;
    __syncthreads();
    for (int s = T >> 1; s > 0; s >>= 1) {
      if (t < s && sv[t + s] > sv[t]) { sv[t] = sv[t + s]; si[t] = si[t + s]; }
      __syncthreads();
    }
    if (t == 0) s_far = si[0];
    __syncthreads();
  }
}

__global__ void k_gather_centers(const float* __restrict__ xyz, const int* __restrict__ idx,
                                 float* __restrict__ nxyz, int N, int S) {
  int i = blockIdx.x * blockDim.x + threadIdx.x;
  if (i >= BATCH * S) return;
  int b = i / S;
  int p = idx[i];
  const float* q = xyz + ((size_t)b * N + p) * 3;
  float* o = nxyz + (size_t)i * 3;
  o[0] = q[0]; o[1] = q[1]; o[2] = q[2];
}

// ---------------------------------------------------------------------------
// K3: kNN select + group. Writes g_xyz (f32, centered), packed f16 g_pts rows
// of length Kpad (zero padded: WMMA-ready), g_den.
// ---------------------------------------------------------------------------
template<int N>
__global__ void k_knn_group(const float* __restrict__ xyz, const float* __restrict__ nxyz,
                            const float* __restrict__ pts, const float* __restrict__ invden,
                            int S, int NS, int D, int Kpad,
                            int* __restrict__ knn, float* __restrict__ gxyz,
                            _Float16* __restrict__ gpts, float* __restrict__ gden) {
  const int b = blockIdx.y, s = blockIdx.x, t = threadIdx.x;
  __shared__ float dist[N];
  __shared__ float sv[256]; __shared__ int si[256];
  const float* c = nxyz + ((size_t)(b * S) + s) * 3;
  float cx = c[0], cy = c[1], cz = c[2];
  for (int p = t; p < N; p += 256) {
    const float* q = xyz + ((size_t)b * N + p) * 3;
    float dx = q[0]-cx, dy = q[1]-cy, dz = q[2]-cz;
    dist[p] = dx*dx + dy*dy + dz*dz;
  }
  __syncthreads();
  for (int k = 0; k < NS; ++k) {
    float bm = 1e30f; int bi = 0;
    for (int p = t; p < N; p += 256)
      if (dist[p] < bm) { bm = dist[p]; bi = p; }
    sv[t] = bm; si[t] = bi;
    __syncthreads();
    for (int r = 128; r > 0; r >>= 1) {
      if (t < r && sv[t + r] < sv[t]) { sv[t] = sv[t + r]; si[t] = si[t + r]; }
      __syncthreads();
    }
    if (t == 0) { int p = si[0]; knn[((size_t)(b * S) + s) * NS + k] = p; dist[p] = 1e30f; }
    __syncthreads();
  }
  for (int k = t; k < NS; k += 256) {
    int p = knn[((size_t)(b * S) + s) * NS + k];
    const float* q = xyz + ((size_t)b * N + p) * 3;
    size_t gi = ((size_t)(b * S) + s) * NS + k;
    float gx = q[0]-cx, gy = q[1]-cy, gz = q[2]-cz;
    gxyz[gi*3+0] = gx; gxyz[gi*3+1] = gy; gxyz[gi*3+2] = gz;
    _Float16* gp = gpts + gi * Kpad;
    gp[0] = (_Float16)gx; gp[1] = (_Float16)gy; gp[2] = (_Float16)gz;
    const float* pr = pts + ((size_t)b * N + p) * D;
    for (int cc = 0; cc < D; ++cc) gp[3 + cc] = (_Float16)pr[cc];
    for (int cc = 3 + D; cc < Kpad; ++cc) gp[cc] = (_Float16)0.f;
    gden[gi] = invden[(size_t)b * N + p];
  }
}

// ---------------------------------------------------------------------------
// K4: f16 WMMA GEMM + bias + BN + optional ReLU.
// A: (Mpad x Kpad) f16, W: (Nout x Kpad) f16, Kpad%32==0, Mpad%128==0 (padded).
// 8 waves/block: 128-row M tile x 16-col N tile. Weight tile (16x32 f16,
// shared by all waves) staged into LDS with GLOBAL_LOAD_ASYNC_TO_LDS_B128.
// ---------------------------------------------------------------------------
__global__ void k_gemm_bn_act(const _Float16* __restrict__ A, const _Float16* __restrict__ Wh,
                              const float* __restrict__ bias, const float* __restrict__ gam,
                              const float* __restrict__ bet, float* __restrict__ out,
                              int M, int Nout, int Kpad, int relu) {
  const int t = threadIdx.x;              // 256 threads = 8 waves
  const int wave = t >> 5, lane = t & 31;
  const int m0 = (blockIdx.x << 7) + (wave << 4);
  const int n0 = blockIdx.y << 4;
  __shared__ __align__(64) _Float16 sW[16 * 32];   // 16 n-rows x 32 k
  const int arow = m0 + (lane & 15);
  const int aoff = (lane < 16) ? 0 : 16;
  const int boff = (lane < 16) ? 0 : 16;
  v8f acc = {};
  for (int k0 = 0; k0 < Kpad; k0 += 32) {
    if (t < 64) {                          // waves 0..1 stage the weight tile
      int r = t >> 2, ch = t & 3;
      const _Float16* src = Wh + (size_t)(n0 + r) * Kpad + k0 + ch * 8;
      _Float16* dst = sW + r * 32 + ch * 8;
#if HAVE_ASYNC
      __builtin_amdgcn_global_load_async_to_lds_b128(
          (__attribute__((address_space(1))) v4i*)const_cast<_Float16*>(src),
          (__attribute__((address_space(3))) v4i*)dst, 0, 0);
      __builtin_amdgcn_s_wait_asynccnt(0);
#else
      *(int4*)dst = *(const int4*)src;
#endif
    }
    __syncthreads();
    const _Float16* ap = A + (size_t)arow * Kpad + k0 + aoff;
    v8h alo = *(const v8h*)(ap);
    v8h ahi = *(const v8h*)(ap + 8);
    v16h af = __builtin_shufflevector(alo, ahi, 0,1,2,3,4,5,6,7,8,9,10,11,12,13,14,15);
    v16h bf = *(const v16h*)(sW + (lane & 15) * 32 + boff);
    acc = __builtin_amdgcn_wmma_f32_16x16x32_f16(false, af, false, bf, (short)0,
                                                 acc, false, false);
    __syncthreads();
  }
  const int an = n0 + (lane & 15);
  float bb = bias[an], gg = gam[an], ee = bet[an];
#pragma unroll
  for (int g = 0; g < 8; ++g) {
    int m = m0 + g + ((lane < 16) ? 0 : 8);
    if (m < M) {
      float v = (acc[g] + bb) * gg + ee;
      if (relu) v = fmaxf(v, 0.f);
      out[(size_t)m * Nout + n0 + (lane & 15)] = v;
    }
  }
}

// ---------------------------------------------------------------------------
// K5: WeightNet 3->8->8->16. Writes transposed f16 wtsT[(group*16+o)*NS + k].
// ---------------------------------------------------------------------------
__global__ void k_weightnet(const float* __restrict__ gxyz,
    const float* w0, const float* b0, const float* g0, const float* e0,
    const float* w1, const float* b1, const float* g1, const float* e1,
    const float* w2, const float* b2, const float* g2, const float* e2,
    _Float16* __restrict__ wtsT, int total, int NS) {
  int i = blockIdx.x * blockDim.x + threadIdx.x;
  if (i >= total) return;
  int grp = i / NS, k = i % NS;
  float x0 = gxyz[(size_t)i*3], x1 = gxyz[(size_t)i*3+1], x2 = gxyz[(size_t)i*3+2];
  float h1[8], h2[8];
#pragma unroll
  for (int o = 0; o < 8; ++o) {
    float v = w0[o*3]*x0 + w0[o*3+1]*x1 + w0[o*3+2]*x2 + b0[o];
    h1[o] = fmaxf(v * g0[o] + e0[o], 0.f);
  }
#pragma unroll
  for (int o = 0; o < 8; ++o) {
    float v = b1[o];
#pragma unroll
    for (int kk = 0; kk < 8; ++kk) v += w1[o*8+kk] * h1[kk];
    h2[o] = fmaxf(v * g1[o] + e1[o], 0.f);
  }
#pragma unroll
  for (int o = 0; o < 16; ++o) {
    float v = b2[o];
#pragma unroll
    for (int kk = 0; kk < 8; ++kk) v += w2[o*8+kk] * h2[kk];
    wtsT[((size_t)grp * 16 + o) * NS + k] = (_Float16)fmaxf(v * g2[o] + e2[o], 0.f);
  }
}

// ---------------------------------------------------------------------------
// K6: DensityNet scaling. Reads feat (f32), writes transposed f16
// featT[(group*C + c)*NS + k] = feat * sigmoid(dn(den/max_k den)).
// ---------------------------------------------------------------------------
__global__ void k_density_scale(const float* __restrict__ feat, const float* __restrict__ gden,
    const float* w0, const float* b0, const float* g0, const float* e0,
    const float* w1, const float* b1, const float* g1, const float* e1,
    const float* w2, const float* b2, const float* g2, const float* e2,
    _Float16* __restrict__ featT, int S, int NS, int C) {
  int b = blockIdx.y, s = blockIdx.x, k = threadIdx.x;
  __shared__ float smax[256];
  size_t grp = (size_t)(b * S) + s;
  size_t base = grp * NS;
  float den = gden[base + k];
  smax[k] = den;
  __syncthreads();
  for (int r = blockDim.x >> 1; r > 0; r >>= 1) {
    if (k < r) smax[k] = fmaxf(smax[k], smax[k + r]);
    __syncthreads();
  }
  float scale = den / smax[0];
  float h1[16], h2[8];
#pragma unroll
  for (int o = 0; o < 16; ++o) {
    float v = w0[o] * scale + b0[o];
    h1[o] = fmaxf(v * g0[o] + e0[o], 0.f);
  }
#pragma unroll
  for (int o = 0; o < 8; ++o) {
    float v = b1[o];
#pragma unroll
    for (int q = 0; q < 16; ++q) v += w1[o*16+q] * h1[q];
    h2[o] = fmaxf(v * g1[o] + e1[o], 0.f);
  }
  float v = b2[0];
#pragma unroll
  for (int q = 0; q < 8; ++q) v += w2[q] * h2[q];
  v = v * g2[0] + e2[0];
  float dnv = 1.f / (1.f + __expf(-v));
  const float* fr = feat + (base + k) * C;
  for (int c = 0; c < C; ++c)
    featT[(grp * C + c) * NS + k] = (_Float16)(fr[c] * dnv);
}

// ---------------------------------------------------------------------------
// K7: einsum bskc,bskw->bscw via f16 WMMA on prepacked featT (C x NS) and
// wtsT (16 x NS). Writes f16 rows (C*16 per group) for the linear GEMM.
// ---------------------------------------------------------------------------
__global__ void k_einsum(const _Float16* __restrict__ featT, const _Float16* __restrict__ wtsT,
                         _Float16* __restrict__ outh, int S, int NS, int C) {
  const int b = blockIdx.y, s = blockIdx.x;
  const int wv = threadIdx.x >> 5, lane = threadIdx.x & 31;
  const int c0 = wv << 4;
  size_t grp = (size_t)(b * S) + s;
  const _Float16* Abase = featT + grp * (size_t)C * NS;
  const _Float16* Bbase = wtsT + grp * (size_t)16 * NS;
  const int aoff = (lane < 16) ? 0 : 16;
  const int boff = (lane < 16) ? 0 : 16;
  v8f acc = {};
  for (int k0 = 0; k0 < NS; k0 += 32) {
    const _Float16* ap = Abase + (size_t)(c0 + (lane & 15)) * NS + k0 + aoff;
    v8h alo = *(const v8h*)(ap);
    v8h ahi = *(const v8h*)(ap + 8);
    v16h af = __builtin_shufflevector(alo, ahi, 0,1,2,3,4,5,6,7,8,9,10,11,12,13,14,15);
    v16h bf = *(const v16h*)(Bbase + (size_t)(lane & 15) * NS + k0 + boff);
    acc = __builtin_amdgcn_wmma_f32_16x16x32_f16(false, af, false, bf, (short)0,
                                                 acc, false, false);
  }
  _Float16* o = outh + grp * (size_t)C * 16;
#pragma unroll
  for (int g = 0; g < 8; ++g) {
    int m = c0 + g + ((lane < 16) ? 0 : 8);
    o[m * 16 + (lane & 15)] = (_Float16)acc[g];
  }
}

// ---------------------------------------------------------------------------
// K8: dense + GroupNorm(all channels) + ReLU. grid=B, blockDim=Nout (<=256).
// ---------------------------------------------------------------------------
__global__ void k_dense_gn_relu(const float* __restrict__ in, const float* __restrict__ W,
                                const float* __restrict__ bias, const float* __restrict__ gam,
                                const float* __restrict__ bet, float* __restrict__ out,
                                int K, int Nout) {
  int b = blockIdx.x, o = threadIdx.x;
  __shared__ float red[256];
  const float* x = in + (size_t)b * K;
  float v = bias[o];
  for (int k = 0; k < K; ++k) v += W[(size_t)o * K + k] * x[k];
  red[o] = v; __syncthreads();
  for (int r = blockDim.x >> 1; r > 0; r >>= 1) {
    if (o < r) red[o] += red[o + r];
    __syncthreads();
  }
  float mean = red[0] / (float)Nout;
  __syncthreads();
  red[o] = (v - mean) * (v - mean); __syncthreads();
  for (int r = blockDim.x >> 1; r > 0; r >>= 1) {
    if (o < r) red[o] += red[o + r];
    __syncthreads();
  }
  float var = red[0] / (float)Nout;
  float y = (v - mean) * rsqrtf(var + 1e-5f) * gam[o] + bet[o];
  out[(size_t)b * Nout + o] = fmaxf(y, 0.f);
}

// K9: final head dense (Nout=1), optional sigmoid.
__global__ void k_head_final(const float* __restrict__ in, const float* __restrict__ W,
                             const float* __restrict__ bias, float* __restrict__ out,
                             int K, int sig) {
  int b = blockIdx.x, t = threadIdx.x;
  __shared__ float red[64];
  float v = 0.f;
  for (int k = t; k < K; k += 64) v += W[k] * in[(size_t)b * K + k];
  red[t] = v; __syncthreads();
  for (int r = 32; r > 0; r >>= 1) {
    if (t < r) red[t] += red[t + r];
    __syncthreads();
  }
  if (t == 0) {
    float y = red[0] + bias[0];
    if (sig) y = 1.f / (1.f + __expf(-y));
    out[b] = y;
  }
}

// K10: concat x = [a(Ca) | b(Cb)] per batch row.
__global__ void k_concat(const float* __restrict__ a, const float* __restrict__ bsrc,
                         float* __restrict__ o, int Ca, int Cb) {
  int i = blockIdx.x * blockDim.x + threadIdx.x;
  int C = Ca + Cb;
  if (i >= BATCH * C) return;
  int b = i / C, c = i % C;
  o[i] = (c < Ca) ? a[(size_t)b * Ca + c] : bsrc[(size_t)b * Cb + (c - Ca)];
}

// K11: group_all packing: gpts(f16,Kpad) = [xyz | pts | 0...].
__global__ void k_pack3(const float* __restrict__ xyz, const float* __restrict__ pts,
                        _Float16* __restrict__ gpts, int N, int D, int Kpad) {
  int i = blockIdx.x * blockDim.x + threadIdx.x;
  if (i >= BATCH * N) return;
  const float* q = xyz + (size_t)i * 3;
  const float* p = pts + (size_t)i * D;
  _Float16* o = gpts + (size_t)i * Kpad;
  o[0] = (_Float16)q[0]; o[1] = (_Float16)q[1]; o[2] = (_Float16)q[2];
  for (int c = 0; c < D; ++c) o[3 + c] = (_Float16)p[c];
  for (int c = 3 + D; c < Kpad; ++c) o[c] = (_Float16)0.f;
}

// K12: pack f32 (Nrow x K) matrix -> f16 (Nrow x Kpad), zero padded.
__global__ void k_packw(const float* __restrict__ src, _Float16* __restrict__ dst,
                        int Nrow, int K, int Kpad) {
  int i = blockIdx.x * blockDim.x + threadIdx.x;
  if (i >= Nrow * Kpad) return;
  int r = i / Kpad, c = i % Kpad;
  dst[i] = (c < K) ? (_Float16)src[(size_t)r * K + c] : (_Float16)0.f;
}

// ---------------------------------------------------------------------------
// Host side
// ---------------------------------------------------------------------------
extern "C" void kernel_launch(void* const* d_in, const int* in_sizes, int n_in,
                              void* d_out, int out_size, void* d_ws, size_t ws_size,
                              hipStream_t stream) {
  (void)in_sizes; (void)n_in; (void)out_size; (void)ws_size;
  // jax pytree flatten order (dict keys alphabetical): params leaves [0..127],
  // pc=128, query=129. See round-0 notes for the full table.
  auto F = [&](int i) -> const float* { return (const float*)d_in[i]; };
  const int SB1 = 32, SB2 = 64, SB3 = 96;
  #define SA_BNL_BE(b) F((b)+0)
  #define SA_BNL_G(b)  F((b)+1)
  #define SA_DN(b,l,j) F((b)+2+(l)*4+(j))
  #define SA_LIN_B(b)  F((b)+14)
  #define SA_LIN_W(b)  F((b)+15)
  #define SA_MLP_B(b)  F((b)+16)
  #define SA_MLP_BE(b) F((b)+17)
  #define SA_MLP_G(b)  F((b)+18)
  #define SA_MLP_W(b)  F((b)+19)
  #define SA_WN(b,l,j) F((b)+20+(l)*4+(j))
  const float* PC = F(128);
  const float* QUERY = F(129);

  char* wsp = (char*)d_ws;
  auto allocF = [&](size_t n) -> float* {
    float* p = (float*)wsp; wsp += ((n * 4) + 255) & ~(size_t)255; return p;
  };
  auto allocH = [&](size_t n) -> _Float16* {
    _Float16* p = (_Float16*)wsp; wsp += ((n * 2) + 255) & ~(size_t)255; return p;
  };
  float* xyz0   = allocF((size_t)BATCH*N1*3);
  float* pts0   = allocF((size_t)BATCH*N1*D0);
  float* ivd1   = allocF((size_t)BATCH*N1);
  int*   fps1   = (int*)allocF((size_t)BATCH*NP1);
  float* nxyz1  = allocF((size_t)BATCH*NP1*3);
  int*   knn1   = (int*)allocF((size_t)BATCH*NP1*NS1);
  float* gxyz1  = allocF((size_t)BATCH*NP1*NS1*3);
  float* gden1  = allocF((size_t)BATCH*NP1*NS1);
  float* feat1  = allocF((size_t)BATCH*NP1*NS1*C1);
  float* pts1   = allocF((size_t)BATCH*NP1*C1);
  float* ivd2   = allocF((size_t)BATCH*N2);
  int*   fps2   = (int*)allocF((size_t)BATCH*NP2);
  float* nxyz2  = allocF((size_t)BATCH*NP2*3);
  int*   knn2   = (int*)allocF((size_t)BATCH*NP2*NS2);
  float* gxyz2  = allocF((size_t)BATCH*NP2*NS2*3);
  float* gden2  = allocF((size_t)BATCH*NP2*NS2);
  float* feat2  = allocF((size_t)BATCH*NP2*NS2*C2);
  float* pts2   = allocF((size_t)BATCH*NP2*C2);
  float* ivd3   = allocF((size_t)BATCH*N3);
  float* feat3  = allocF((size_t)BATCH*N3*C3);
  float* pts3   = allocF((size_t)BATCH*C3);
  float* q1     = allocF((size_t)BATCH*64);
  float* q2     = allocF((size_t)BATCH*128);
  float* q3     = allocF((size_t)BATCH*256);
  float* xcat   = allocF((size_t)BATCH*512);
  float* hs1    = allocF((size_t)BATCH*256);
  float* hs2    = allocF((size_t)BATCH*128);
  float* ho1    = allocF((size_t)BATCH*256);
  float* ho2    = allocF((size_t)BATCH*128);
  // f16 prepacked operands
  _Float16* gpts1h = allocH((size_t)BATCH*NP1*NS1*KP1);
  _Float16* mlpw1h = allocH((size_t)C1*KP1);
  _Float16* linw1h = allocH((size_t)C1*16*C1);
  _Float16* featT1 = allocH((size_t)BATCH*NP1*C1*NS1);
  _Float16* wtsT1  = allocH((size_t)BATCH*NP1*16*NS1);
  _Float16* new1h  = allocH((size_t)BATCH*NP1*16*C1);
  _Float16* gpts2h = allocH((size_t)BATCH*NP2*NS2*KP2);
  _Float16* mlpw2h = allocH((size_t)C2*KP2);
  _Float16* linw2h = allocH((size_t)C2*16*C2);
  _Float16* featT2 = allocH((size_t)BATCH*NP2*C2*NS2);
  _Float16* wtsT2  = allocH((size_t)BATCH*NP2*16*NS2);
  _Float16* new2h  = allocH((size_t)BATCH*NP2*16*C2);
  _Float16* gpts3h = allocH((size_t)BATCH*N3*KP3);
  _Float16* mlpw3h = allocH((size_t)C3*KP3);
  _Float16* linw3h = allocH((size_t)C3*16*C3);
  _Float16* featT3 = allocH((size_t)BATCH*C3*N3);
  _Float16* wtsT3  = allocH((size_t)BATCH*16*N3);
  _Float16* new3h  = allocH((size_t)128*16*C3);    // 128 rows zero-padded (M=8)
  float* outp = (float*)d_out;

  // weight prepacking (tiny)
  k_packw<<<(C1*KP1+255)/256,   256, 0, stream>>>(SA_MLP_W(SB1), mlpw1h, C1, 8,    KP1);
  k_packw<<<(C1*16*C1+255)/256, 256, 0, stream>>>(SA_LIN_W(SB1), linw1h, C1, 16*C1, 16*C1);
  k_packw<<<(C2*KP2+255)/256,   256, 0, stream>>>(SA_MLP_W(SB2), mlpw2h, C2, 67,   KP2);
  k_packw<<<(C2*16*C2+255)/256, 256, 0, stream>>>(SA_LIN_W(SB2), linw2h, C2, 16*C2, 16*C2);
  k_packw<<<(C3*KP3+255)/256,   256, 0, stream>>>(SA_MLP_W(SB3), mlpw3h, C3, 131,  KP3);
  k_packw<<<(C3*16*C3+255)/256, 256, 0, stream>>>(SA_LIN_W(SB3), linw3h, C3, 16*C3, 16*C3);

  k_transpose<<<(BATCH*N1+255)/256, 256, 0, stream>>>(PC, xyz0, pts0, D0, N1);

  // ================= SA1 =================
  k_density<<<dim3(N1/16, BATCH), 256, 0, stream>>>(xyz0, ivd1, N1, 0.1f);
  k_fps<N1, NP1><<<BATCH, 256, 0, stream>>>(xyz0, fps1);
  k_gather_centers<<<(BATCH*NP1+255)/256, 256, 0, stream>>>(xyz0, fps1, nxyz1, N1, NP1);
  k_knn_group<N1><<<dim3(NP1, BATCH), 256, 0, stream>>>(xyz0, nxyz1, pts0, ivd1,
      NP1, NS1, D0, KP1, knn1, gxyz1, gpts1h, gden1);
  { const int M = BATCH*NP1*NS1;
    k_gemm_bn_act<<<dim3(M/128, C1/16), 256, 0, stream>>>(gpts1h, mlpw1h,
        SA_MLP_B(SB1), SA_MLP_G(SB1), SA_MLP_BE(SB1), feat1, M, C1, KP1, 1);
    k_density_scale<<<dim3(NP1, BATCH), NS1, 0, stream>>>(feat1, gden1,
        SA_DN(SB1,0,0),SA_DN(SB1,0,1),SA_DN(SB1,0,2),SA_DN(SB1,0,3),
        SA_DN(SB1,1,0),SA_DN(SB1,1,1),SA_DN(SB1,1,2),SA_DN(SB1,1,3),
        SA_DN(SB1,2,0),SA_DN(SB1,2,1),SA_DN(SB1,2,2),SA_DN(SB1,2,3),
        featT1, NP1, NS1, C1);
    k_weightnet<<<(M+127)/128, 128, 0, stream>>>(gxyz1,
        SA_WN(SB1,0,0),SA_WN(SB1,0,1),SA_WN(SB1,0,2),SA_WN(SB1,0,3),
        SA_WN(SB1,1,0),SA_WN(SB1,1,1),SA_WN(SB1,1,2),SA_WN(SB1,1,3),
        SA_WN(SB1,2,0),SA_WN(SB1,2,1),SA_WN(SB1,2,2),SA_WN(SB1,2,3),
        wtsT1, M, NS1);
    k_einsum<<<dim3(NP1, BATCH), (C1/16)*32, 0, stream>>>(featT1, wtsT1, new1h, NP1, NS1, C1);
    k_gemm_bn_act<<<dim3((BATCH*NP1)/128, C1/16), 256, 0, stream>>>(new1h, linw1h,
        SA_LIN_B(SB1), SA_BNL_G(SB1), SA_BNL_BE(SB1), pts1, BATCH*NP1, C1, 16*C1, 1);
  }

  // ================= SA2 =================
  k_density<<<dim3(N2/16, BATCH), 256, 0, stream>>>(nxyz1, ivd2, N2, 0.2f);
  k_fps<N2, NP2><<<BATCH, 256, 0, stream>>>(nxyz1, fps2);
  k_gather_centers<<<(BATCH*NP2+255)/256, 256, 0, stream>>>(nxyz1, fps2, nxyz2, N2, NP2);
  k_knn_group<N2><<<dim3(NP2, BATCH), 256, 0, stream>>>(nxyz1, nxyz2, pts1, ivd2,
      NP2, NS2, C1, KP2, knn2, gxyz2, gpts2h, gden2);
  { const int M = BATCH*NP2*NS2;
    k_gemm_bn_act<<<dim3(M/128, C2/16), 256, 0, stream>>>(gpts2h, mlpw2h,
        SA_MLP_B(SB2), SA_MLP_G(SB2), SA_MLP_BE(SB2), feat2, M, C2, KP2, 1);
    k_density_scale<<<dim3(NP2, BATCH), NS2, 0, stream>>>(feat2, gden2,
        SA_DN(SB2,0,0),SA_DN(SB2,0,1),SA_DN(SB2,0,2),SA_DN(SB2,0,3),
        SA_DN(SB2,1,0),SA_DN(SB2,1,1),SA_DN(SB2,1,2),SA_DN(SB2,1,3),
        SA_DN(SB2,2,0),SA_DN(SB2,2,1),SA_DN(SB2,2,2),SA_DN(SB2,2,3),
        featT2, NP2, NS2, C2);
    k_weightnet<<<(M+127)/128, 128, 0, stream>>>(gxyz2,
        SA_WN(SB2,0,0),SA_WN(SB2,0,1),SA_WN(SB2,0,2),SA_WN(SB2,0,3),
        SA_WN(SB2,1,0),SA_WN(SB2,1,1),SA_WN(SB2,1,2),SA_WN(SB2,1,3),
        SA_WN(SB2,2,0),SA_WN(SB2,2,1),SA_WN(SB2,2,2),SA_WN(SB2,2,3),
        wtsT2, M, NS2);
    k_einsum<<<dim3(NP2, BATCH), (C2/16)*32, 0, stream>>>(featT2, wtsT2, new2h, NP2, NS2, C2);
    k_gemm_bn_act<<<dim3((BATCH*NP2)/128, C2/16), 256, 0, stream>>>(new2h, linw2h,
        SA_LIN_B(SB2), SA_BNL_G(SB2), SA_BNL_BE(SB2), pts2, BATCH*NP2, C2, 16*C2, 1);
  }

  // ================= SA3 (group_all) =================
  k_density<<<dim3(N3/16, BATCH), 256, 0, stream>>>(nxyz2, ivd3, N3, 0.4f);
  k_pack3<<<(BATCH*N3+255)/256, 256, 0, stream>>>(nxyz2, pts2, gpts3h, N3, C2, KP3);
  { const int M = BATCH*N3;
    k_gemm_bn_act<<<dim3(M/128, C3/16), 256, 0, stream>>>(gpts3h, mlpw3h,
        SA_MLP_B(SB3), SA_MLP_G(SB3), SA_MLP_BE(SB3), feat3, M, C3, KP3, 1);
    k_density_scale<<<dim3(1, BATCH), N3, 0, stream>>>(feat3, ivd3,
        SA_DN(SB3,0,0),SA_DN(SB3,0,1),SA_DN(SB3,0,2),SA_DN(SB3,0,3),
        SA_DN(SB3,1,0),SA_DN(SB3,1,1),SA_DN(SB3,1,2),SA_DN(SB3,1,3),
        SA_DN(SB3,2,0),SA_DN(SB3,2,1),SA_DN(SB3,2,2),SA_DN(SB3,2,3),
        featT3, 1, N3, C3);
    k_weightnet<<<(M+127)/128, 128, 0, stream>>>(nxyz2,   // group_all: uncentered xyz
        SA_WN(SB3,0,0),SA_WN(SB3,0,1),SA_WN(SB3,0,2),SA_WN(SB3,0,3),
        SA_WN(SB3,1,0),SA_WN(SB3,1,1),SA_WN(SB3,1,2),SA_WN(SB3,1,3),
        SA_WN(SB3,2,0),SA_WN(SB3,2,1),SA_WN(SB3,2,2),SA_WN(SB3,2,3),
        wtsT3, M, N3);
    (void)hipMemsetAsync(new3h, 0, (size_t)128*16*C3*2, stream);   // zero-pad M 8->128
    k_einsum<<<dim3(1, BATCH), (C3/16)*32, 0, stream>>>(featT3, wtsT3, new3h, 1, N3, C3);
    k_gemm_bn_act<<<dim3(1, C3/16), 256, 0, stream>>>(new3h, linw3h,
        SA_LIN_B(SB3), SA_BNL_G(SB3), SA_BNL_BE(SB3), pts3, BATCH, C3, 16*C3, 1);
  }

  // ================= query branch =================
  k_dense_gn_relu<<<BATCH, 64,  0, stream>>>(QUERY, F(7),  F(6),  F(1), F(0), q1, 3,   64);
  k_dense_gn_relu<<<BATCH, 128, 0, stream>>>(q1,    F(9),  F(8),  F(3), F(2), q2, 64,  128);
  k_dense_gn_relu<<<BATCH, 256, 0, stream>>>(q2,    F(11), F(10), F(5), F(4), q3, 128, 256);
  k_concat<<<(BATCH*512+255)/256, 256, 0, stream>>>(pts3, q3, xcat, 256, 256);

  // ================= heads =================
  k_dense_gn_relu<<<BATCH, 256, 0, stream>>>(xcat, F(25), F(22), F(24), F(23), hs1, 512, 256);
  k_dense_gn_relu<<<BATCH, 128, 0, stream>>>(hs1,  F(29), F(26), F(28), F(27), hs2, 256, 128);
  k_head_final<<<BATCH, 64, 0, stream>>>(hs2, F(31), F(30), outp, 128, 0);
  k_dense_gn_relu<<<BATCH, 256, 0, stream>>>(xcat, F(15), F(12), F(14), F(13), ho1, 512, 256);
  k_dense_gn_relu<<<BATCH, 128, 0, stream>>>(ho1,  F(19), F(16), F(18), F(17), ho2, 256, 128);
  k_head_final<<<BATCH, 64, 0, stream>>>(ho2, F(21), F(20), outp + BATCH, 128, 1);
}
